// AttentionDecoder_60378650247765
// MI455X (gfx1250) — matrix operand order repaired and checked by reading, not compiled, not run on hardware
//
#include <hip/hip_runtime.h>

typedef __attribute__((ext_vector_type(16))) _Float16 v16h;
typedef __attribute__((ext_vector_type(8)))  _Float16 v8h;
typedef __attribute__((ext_vector_type(4)))  _Float16 v4h;
typedef __attribute__((ext_vector_type(2)))  _Float16 v2h;
typedef __attribute__((ext_vector_type(8)))  float    v8f;
typedef __attribute__((ext_vector_type(4)))  int      v4i;

#define BDIM 128
#define HDIM 512
#define DDIM 512
#define LDIM 2048
#define NG   2048   // 4*H
#define KK   1024   // D + H
#define TSTEPS 10
#define ATT_SPLIT 4
#define ATT_CHUNK 32   // rows per chunk, double buffered

// ---- CDNA5 async global->LDS copy (ASYNCcnt) with safe fallback -------------

#if defined(__AMDGCN__) && __has_builtin(__builtin_amdgcn_global_load_async_to_lds_b128)
typedef __attribute__((address_space(1))) v4i as1_v4i;   // global
typedef __attribute__((address_space(3))) v4i as3_v4i;   // LDS
__device__ __forceinline__ void async_cp16(const void* g, void* l) {
  __builtin_amdgcn_global_load_async_to_lds_b128(
      (as1_v4i*)g, (as3_v4i*)l, /*imm offset*/0, /*cpol*/0);
}
#define WAIT_ASYNC_16() asm volatile("s_wait_asynccnt 0x10" ::: "memory")
#define WAIT_ASYNC_8()  asm volatile("s_wait_asynccnt 0x8" ::: "memory")
#define WAIT_ASYNC_0()  asm volatile("s_wait_asynccnt 0x0" ::: "memory")
#else
__device__ __forceinline__ void async_cp16(const void* g, void* l) {
  *(float4*)l = *(const float4*)g;   // sync fallback; barrier provides ordering
}
#define WAIT_ASYNC_16() do {} while (0)
#define WAIT_ASYNC_8()  do {} while (0)
#define WAIT_ASYNC_0()  do {} while (0)
#endif

// ---------------------------------------------------------------- init / prep

__global__ __launch_bounds__(256) void init_kernel(
    const float* __restrict__ batch, const float* __restrict__ h0,
    const float* __restrict__ s0, float* __restrict__ x,
    float* __restrict__ h, float* __restrict__ s) {
  int i = blockIdx.x * blockDim.x + threadIdx.x;
  if (i < BDIM * HDIM) { x[i] = batch[i]; h[i] = h0[i]; s[i] = s0[i]; }
}

// Wt[n][k] = (k<512 ? W_ih[k][n] : W_hh[k-512][n]) as f16, row-major (NG x KK)
__global__ __launch_bounds__(256) void wconv_kernel(
    const float* __restrict__ W_ih, const float* __restrict__ W_hh,
    _Float16* __restrict__ Wt) {
  int idx = blockIdx.x * blockDim.x + threadIdx.x;   // NG*KK = 2M
  if (idx >= NG * KK) return;
  int k = idx >> 11;          // / NG
  int n = idx & (NG - 1);
  float v = (k < DDIM) ? W_ih[(size_t)k * NG + n] : W_hh[(size_t)(k - DDIM) * NG + n];
  Wt[(size_t)n * KK + k] = (_Float16)v;
}

// hid fp32 -> f16 (halves all subsequent attention HBM traffic)
__global__ __launch_bounds__(256) void hconv_kernel(
    const float* __restrict__ hid, _Float16* __restrict__ hid16) {
  size_t i = ((size_t)blockIdx.x * 256 + threadIdx.x) * 4;
  if (i >= (size_t)BDIM * LDIM * HDIM) return;
  float4 v = *(const float4*)(hid + i);
  v4h o;
  o[0] = (_Float16)v.x; o[1] = (_Float16)v.y; o[2] = (_Float16)v.z; o[3] = (_Float16)v.w;
  *(v4h*)(hid16 + i) = o;
}

// ---------- flash attention, f16 stream (L split 4, async double-buffered LDS)

__global__ __launch_bounds__(256) void attn_partial_f16(
    const _Float16* __restrict__ hid16, const float* __restrict__ s_state,
    float* __restrict__ pm, float* __restrict__ pZ, float* __restrict__ pctx) {
  extern __shared__ float sm[];
  float* s_l    = sm;            // 512
  float* ctx    = sm + 512;      // 512
  float* scores = sm + 1024;     // 32 (+pad)
  float* red    = sm + 1088;     // 256
  float* scal   = sm + 1344;     // [0]=ss [1]=m [2]=Z [3]=scale (+pad to 1352)
  _Float16* bufs = (_Float16*)(sm + 1352);   // 2 x (32*512) halves = 64KB

  int tid  = threadIdx.x;
  int lane = tid & 31, wave = tid >> 5;
  int b = blockIdx.x >> 2, part = blockIdx.x & 3;
  const int rows = LDIM / ATT_SPLIT;            // 512
  const int nch  = rows / ATT_CHUNK;            // 16
  const _Float16* base = hid16 + ((size_t)b * LDIM + (size_t)part * rows) * HDIM;

  float ssp = 0.f;
  for (int i = tid; i < HDIM; i += 256) {
    float v = s_state[b * HDIM + i];
    s_l[i] = v; ctx[i] = 0.f; ssp += v * v;
  }
  red[tid] = ssp; __syncthreads();
  for (int s = 128; s > 0; s >>= 1) { if (tid < s) red[tid] += red[tid + s]; __syncthreads(); }
  if (tid == 0) { scal[0] = red[0]; scal[1] = -3.4e38f; scal[2] = 0.f; }
  __syncthreads();

  // prologue: chunk 0 -> buffer 0 (8 b128 async ops per wave: 32KB chunk)
  {
    #pragma unroll
    for (int j = 0; j < 8; ++j) {
      int i = (tid + 256 * j) * 8;   // 8 halves = 16B per op
      async_cp16(base + i, bufs + i);
    }
  }

  for (int c = 0; c < nch; ++c) {
    if (c + 1 < nch) {   // issue next chunk, then wait for current (in-order)
      const _Float16* src = base + (size_t)(c + 1) * ATT_CHUNK * HDIM;
      _Float16* dst = bufs + ((c + 1) & 1) * (ATT_CHUNK * HDIM);
      #pragma unroll
      for (int j = 0; j < 8; ++j) {
        int i = (tid + 256 * j) * 8;
        async_cp16(src + i, dst + i);
      }
      WAIT_ASYNC_8();
    } else {
      WAIT_ASYNC_0();
    }
    __syncthreads();
    const _Float16* ch = bufs + (c & 1) * (ATT_CHUNK * HDIM);

    // scores + on-the-fly row sum-of-squares: 8 waves x 4 rows
    #pragma unroll
    for (int rr = 0; rr < ATT_CHUNK / 8; ++rr) {
      int row = wave * (ATT_CHUNK / 8) + rr;
      const _Float16* cr = ch + row * HDIM;
      float d = 0.f, q = 0.f;
      #pragma unroll
      for (int t8 = 0; t8 < 2; ++t8) {
        int kbase = lane * 8 + t8 * 256;
        v8h hv = *(const v8h*)(cr + kbase);
        #pragma unroll
        for (int j = 0; j < 8; ++j) {
          float f = (float)hv[j];
          d += f * s_l[kbase + j];
          q += f * f;
        }
      }
      #pragma unroll
      for (int o = 16; o; o >>= 1) { d += __shfl_down(d, o, 32); q += __shfl_down(q, o, 32); }
      if (lane == 0) scores[row] = d / (sqrtf(q * scal[0]) + 1e-8f);
    }
    __syncthreads();
    if (tid == 0) {
      float cm = scores[0];
      for (int i = 1; i < ATT_CHUNK; ++i) cm = fmaxf(cm, scores[i]);
      float mnew = fmaxf(scal[1], cm);
      scal[3] = __expf(scal[1] - mnew);
      scal[1] = mnew;
    }
    __syncthreads();
    float scaleF = scal[3];
    if (tid < ATT_CHUNK) scores[tid] = __expf(scores[tid] - scal[1]);
    __syncthreads();
    if (tid == 0) {
      float zs = 0.f;
      for (int i = 0; i < ATT_CHUNK; ++i) zs += scores[i];
      scal[2] = scal[2] * scaleF + zs;
    }
    // context accumulate: thread owns contiguous pair h = {2*tid, 2*tid+1}
    float ax = 0.f, ay = 0.f;
    #pragma unroll 4
    for (int row = 0; row < ATT_CHUNK; ++row) {
      float p = scores[row];
      v2h v = *(const v2h*)(ch + row * HDIM + 2 * tid);
      ax += p * (float)v[0]; ay += p * (float)v[1];
    }
    float2* ctx2 = (float2*)ctx;
    float2 o2 = ctx2[tid];
    o2.x = o2.x * scaleF + ax;
    o2.y = o2.y * scaleF + ay;
    ctx2[tid] = o2;
    __syncthreads();
  }
  int pi = b * ATT_SPLIT + part;
  if (tid == 0) { pm[pi] = scal[1]; pZ[pi] = scal[2]; }
  ((float2*)(pctx + (size_t)pi * HDIM))[tid] = ((float2*)ctx)[tid];
}

// ---------- flash attention, fp32 stream (fallback if workspace too small)

__global__ __launch_bounds__(256) void attn_partial_f32(
    const float* __restrict__ hid, const float* __restrict__ s_state,
    float* __restrict__ pm, float* __restrict__ pZ, float* __restrict__ pctx) {
  extern __shared__ float sm[];
  float* s_l    = sm;
  float* ctx    = sm + 512;
  float* scores = sm + 1024;
  float* red    = sm + 1088;
  float* scal   = sm + 1344;
  float* bufs   = sm + 1352;     // 2 x (32*512) floats = 128KB

  int tid  = threadIdx.x;
  int lane = tid & 31, wave = tid >> 5;
  int b = blockIdx.x >> 2, part = blockIdx.x & 3;
  const int rows = LDIM / ATT_SPLIT;
  const int nch  = rows / ATT_CHUNK;
  const float* base = hid + ((size_t)b * LDIM + (size_t)part * rows) * HDIM;

  float ssp = 0.f;
  for (int i = tid; i < HDIM; i += 256) {
    float v = s_state[b * HDIM + i];
    s_l[i] = v; ctx[i] = 0.f; ssp += v * v;
  }
  red[tid] = ssp; __syncthreads();
  for (int s = 128; s > 0; s >>= 1) { if (tid < s) red[tid] += red[tid + s]; __syncthreads(); }
  if (tid == 0) { scal[0] = red[0]; scal[1] = -3.4e38f; scal[2] = 0.f; }
  __syncthreads();

  {
    #pragma unroll
    for (int j = 0; j < 16; ++j) {
      int i = (tid + 256 * j) * 4;
      async_cp16(base + i, bufs + i);
    }
  }
  for (int c = 0; c < nch; ++c) {
    if (c + 1 < nch) {
      const float* src = base + (size_t)(c + 1) * ATT_CHUNK * HDIM;
      float* dst = bufs + ((c + 1) & 1) * (ATT_CHUNK * HDIM);
      #pragma unroll
      for (int j = 0; j < 16; ++j) {
        int i = (tid + 256 * j) * 4;
        async_cp16(src + i, dst + i);
      }
      WAIT_ASYNC_16();
    } else {
      WAIT_ASYNC_0();
    }
    __syncthreads();
    const float* ch = bufs + (c & 1) * (ATT_CHUNK * HDIM);

    #pragma unroll
    for (int rr = 0; rr < ATT_CHUNK / 8; ++rr) {
      int row = wave * (ATT_CHUNK / 8) + rr;
      const float* cr = ch + row * HDIM;
      float d = 0.f, q = 0.f;
      #pragma unroll
      for (int t4 = 0; t4 < 4; ++t4) {
        float4 v  = *(const float4*)(cr  + lane * 4 + t4 * 128);
        float4 s4 = *(const float4*)(s_l + lane * 4 + t4 * 128);
        d += v.x * s4.x + v.y * s4.y + v.z * s4.z + v.w * s4.w;
        q += v.x * v.x + v.y * v.y + v.z * v.z + v.w * v.w;
      }
      #pragma unroll
      for (int o = 16; o; o >>= 1) { d += __shfl_down(d, o, 32); q += __shfl_down(q, o, 32); }
      if (lane == 0) scores[row] = d / (sqrtf(q * scal[0]) + 1e-8f);
    }
    __syncthreads();
    if (tid == 0) {
      float cm = scores[0];
      for (int i = 1; i < ATT_CHUNK; ++i) cm = fmaxf(cm, scores[i]);
      float mnew = fmaxf(scal[1], cm);
      scal[3] = __expf(scal[1] - mnew);
      scal[1] = mnew;
    }
    __syncthreads();
    float scaleF = scal[3];
    if (tid < ATT_CHUNK) scores[tid] = __expf(scores[tid] - scal[1]);
    __syncthreads();
    if (tid == 0) {
      float zs = 0.f;
      for (int i = 0; i < ATT_CHUNK; ++i) zs += scores[i];
      scal[2] = scal[2] * scaleF + zs;
    }
    float ax = 0.f, ay = 0.f;
    #pragma unroll 4
    for (int row = 0; row < ATT_CHUNK; ++row) {
      float p = scores[row];
      float2 v = *(const float2*)(ch + row * HDIM + 2 * tid);
      ax += p * v.x; ay += p * v.y;
    }
    float2* ctx2 = (float2*)ctx;
    float2 o2 = ctx2[tid];
    o2.x = o2.x * scaleF + ax;
    o2.y = o2.y * scaleF + ay;
    ctx2[tid] = o2;
    __syncthreads();
  }
  int pi = b * ATT_SPLIT + part;
  if (tid == 0) { pm[pi] = scal[1]; pZ[pi] = scal[2]; }
  ((float2*)(pctx + (size_t)pi * HDIM))[tid] = ((float2*)ctx)[tid];
}

__global__ __launch_bounds__(256) void attn_combine(
    const float* __restrict__ pm, const float* __restrict__ pZ,
    const float* __restrict__ pctx, float* __restrict__ s_state) {
  __shared__ float w[ATT_SPLIT];
  __shared__ float invZ;
  int b = blockIdx.x, tid = threadIdx.x;
  if (tid == 0) {
    float M = pm[b * 4];
    for (int i = 1; i < 4; ++i) M = fmaxf(M, pm[b * 4 + i]);
    float Z = 0.f;
    for (int i = 0; i < 4; ++i) { float e = __expf(pm[b * 4 + i] - M); w[i] = e; Z += pZ[b * 4 + i] * e; }
    invZ = 1.f / Z;
  }
  __syncthreads();
  for (int h = tid; h < HDIM; h += 256) {
    float cx = 0.f;
    #pragma unroll
    for (int i = 0; i < 4; ++i) cx += pctx[((size_t)b * 4 + i) * HDIM + h] * w[i];
    s_state[b * HDIM + h] += cx * invZ;
  }
}

// -------------------------------------------- LSTM gates GEMM via v_wmma f16

__global__ __launch_bounds__(256) void gemm_gates_wmma(
    const float* __restrict__ x, const float* __restrict__ h,
    const _Float16* __restrict__ Wt, const float* __restrict__ bias,
    float* __restrict__ gates) {
  int wave = threadIdx.x >> 5;
  int lane = threadIdx.x & 31;
  int tile = blockIdx.x * 8 + wave;          // 1024 tiles
  const int ntn = NG / 16;                   // 128
  int m0 = (tile / ntn) * 16;
  int n0 = (tile % ntn) * 16;
  int r  = lane & 15;
  int hi = lane >> 4;

  const float*    arow_x = x  + (size_t)(m0 + r) * DDIM;
  const float*    arow_h = h  + (size_t)(m0 + r) * HDIM;
  const _Float16* brow   = Wt + (size_t)(n0 + r) * KK;

  v8f acc = {};
  for (int kb = 0; kb < KK; kb += 32) {
    int k1 = kb + hi * 8;
    int k2 = kb + 16 + hi * 8;
    const float* s1 = (k1 < DDIM) ? (arow_x + k1) : (arow_h + (k1 - DDIM));
    const float* s2 = (k2 < DDIM) ? (arow_x + k2) : (arow_h + (k2 - DDIM));
    float4 f0 = *(const float4*)(s1);
    float4 f1 = *(const float4*)(s1 + 4);
    float4 f2 = *(const float4*)(s2);
    float4 f3 = *(const float4*)(s2 + 4);
    v16h a;
    a[0]  = (_Float16)f0.x; a[1]  = (_Float16)f0.y; a[2]  = (_Float16)f0.z; a[3]  = (_Float16)f0.w;
    a[4]  = (_Float16)f1.x; a[5]  = (_Float16)f1.y; a[6]  = (_Float16)f1.z; a[7]  = (_Float16)f1.w;
    a[8]  = (_Float16)f2.x; a[9]  = (_Float16)f2.y; a[10] = (_Float16)f2.z; a[11] = (_Float16)f2.w;
    a[12] = (_Float16)f3.x; a[13] = (_Float16)f3.y; a[14] = (_Float16)f3.z; a[15] = (_Float16)f3.w;
    v8h b0 = *(const v8h*)(brow + kb + hi * 8);
    v8h b1 = *(const v8h*)(brow + kb + 16 + hi * 8);
    v16h bfrag = __builtin_shufflevector(b0, b1, 0, 1, 2, 3, 4, 5, 6, 7, 8, 9, 10, 11, 12, 13, 14, 15);
    acc = __builtin_amdgcn_wmma_f32_16x16x32_f16(
        false, a, false, bfrag, (short)0, acc, false, false);
  }
  float bn = bias[n0 + r];
  #pragma unroll
  for (int v = 0; v < 8; ++v) {
    int m = m0 + hi * 8 + v;
    gates[(size_t)m * NG + n0 + r] = acc[v] + bn;
  }
}

// ------------------------------------------------------- elementwise LSTM cell

__global__ __launch_bounds__(256) void lstm_elem(
    const float* __restrict__ gates, float* __restrict__ s_state,
    float* __restrict__ hbuf) {
  int idx = blockIdx.x * blockDim.x + threadIdx.x;
  if (idx >= BDIM * HDIM) return;
  int b = idx >> 9, j = idx & 511;
  const float* g = gates + (size_t)b * NG;
  float ig = g[j], fg = g[HDIM + j], gg = g[2 * HDIM + j], og = g[3 * HDIM + j];
  float s  = s_state[idx];
  float si = 1.f / (1.f + __expf(-ig));
  float sf = 1.f / (1.f + __expf(-fg));
  float so = 1.f / (1.f + __expf(-og));
  float c  = sf * s + si * tanhf(gg);
  s_state[idx] = c;
  hbuf[idx]    = so * tanhf(c);
}

// ------------------------------------------------------------------- MLP head

__global__ __launch_bounds__(64) void mlp1_kernel(
    const float* __restrict__ h, const float* __restrict__ W1,
    const float* __restrict__ b1, float* __restrict__ a1) {
  __shared__ float hr[HDIM];
  int b = blockIdx.x, n = threadIdx.x;
  for (int i = n; i < HDIM; i += 64) hr[i] = h[(size_t)b * HDIM + i];
  __syncthreads();
  float acc = b1[n];
  for (int k = 0; k < HDIM; ++k) acc += hr[k] * W1[(size_t)k * 64 + n];
  a1[b * 64 + n] = acc > 0.f ? acc : 0.01f * acc;   // leaky_relu(0.01)
}

__global__ __launch_bounds__(256) void mlp2_kernel(
    const float* __restrict__ a1, const float* __restrict__ W2,
    const float* __restrict__ b2, float* __restrict__ x,
    float* __restrict__ out, int t) {
  __shared__ float ar[64];
  int b = blockIdx.x, tid = threadIdx.x;
  if (tid < 64) ar[tid] = a1[b * 64 + tid];
  __syncthreads();
  for (int d = tid; d < DDIM; d += 256) {
    float acc = b2[d];
    #pragma unroll 8
    for (int k = 0; k < 64; ++k) acc += ar[k] * W2[(size_t)k * DDIM + d];
    x[(size_t)b * DDIM + d] = acc;
    int j = t * DDIM + d;
    out[(size_t)b * (TSTEPS * DDIM) + (size_t)(j % TSTEPS) * DDIM + (j / TSTEPS)] = acc;
  }
}

// --------------------------------------------------------------------- driver

extern "C" void kernel_launch(void* const* d_in, const int* in_sizes, int n_in,
                              void* d_out, int out_size, void* d_ws, size_t ws_size,
                              hipStream_t stream) {
  const float* batch = (const float*)d_in[0];
  const float* hid   = (const float*)d_in[1];
  const float* h0    = (const float*)d_in[2];
  const float* s0    = (const float*)d_in[3];
  const float* W_ih  = (const float*)d_in[4];
  const float* W_hh  = (const float*)d_in[5];
  const float* b_l   = (const float*)d_in[6];
  const float* W1    = (const float*)d_in[7];
  const float* b1    = (const float*)d_in[8];
  const float* W2    = (const float*)d_in[9];
  const float* b2    = (const float*)d_in[10];
  float* out = (float*)d_out;

  char* ws = (char*)d_ws;
  size_t off = 0;
  auto alloc = [&](size_t bytes) -> void* {
    void* p = ws + off;
    off = (off + bytes + 255) & ~(size_t)255;
    return p;
  };
  float*    xbuf   = (float*)alloc((size_t)BDIM * DDIM * 4);
  float*    hbuf   = (float*)alloc((size_t)BDIM * HDIM * 4);
  float*    sstate = (float*)alloc((size_t)BDIM * HDIM * 4);
  float*    gates  = (float*)alloc((size_t)BDIM * NG * 4);
  float*    a1     = (float*)alloc((size_t)BDIM * 64 * 4);
  float*    pm     = (float*)alloc((size_t)BDIM * ATT_SPLIT * 4);
  float*    pZ     = (float*)alloc((size_t)BDIM * ATT_SPLIT * 4);
  float*    pctx   = (float*)alloc((size_t)BDIM * ATT_SPLIT * HDIM * 4);
  _Float16* Wt     = (_Float16*)alloc((size_t)NG * KK * 2);
  _Float16* hid16  = (_Float16*)alloc((size_t)BDIM * LDIM * HDIM * 2);  // 256MB
  const bool use_f16_stream = (off <= ws_size);
  (void)in_sizes; (void)n_in; (void)out_size;

  init_kernel<<<(BDIM * HDIM) / 256, 256, 0, stream>>>(batch, h0, s0, xbuf, hbuf, sstate);
  wconv_kernel<<<(NG * KK) / 256, 256, 0, stream>>>(W_ih, W_hh, Wt);
  if (use_f16_stream)
    hconv_kernel<<<(int)(((size_t)BDIM * LDIM * HDIM / 4) / 256), 256, 0, stream>>>(hid, hid16);

  const size_t lds16 = 1352 * 4 + 2 * ATT_CHUNK * HDIM * 2;   // ~70.9KB -> 4 WG/WGP
  const size_t lds32 = 1352 * 4 + 2 * ATT_CHUNK * HDIM * 4;   // ~136.5KB -> 2 WG/WGP
  for (int t = 0; t < TSTEPS; ++t) {
    if (use_f16_stream)
      attn_partial_f16<<<BDIM * ATT_SPLIT, 256, lds16, stream>>>(hid16, sstate, pm, pZ, pctx);
    else
      attn_partial_f32<<<BDIM * ATT_SPLIT, 256, lds32, stream>>>(hid, sstate, pm, pZ, pctx);
    attn_combine<<<BDIM, 256, 0, stream>>>(pm, pZ, pctx, sstate);
    gemm_gates_wmma<<<(8 * (NG / 16)) / 8, 256, 0, stream>>>(xbuf, hbuf, Wt, b_l, gates);
    lstm_elem<<<(BDIM * HDIM) / 256, 256, 0, stream>>>(gates, sstate, hbuf);
    mlp1_kernel<<<BDIM, 64, 0, stream>>>(hbuf, W1, b1, a1);
    mlp2_kernel<<<BDIM, 256, 0, stream>>>(a1, W2, b2, xbuf, out, t);
  }
}
